// Our_net_76940044141068
// MI455X (gfx1250) — compile-verified, matrix operand order checked by reading, not run on hardware
//
#include <hip/hip_runtime.h>

// ---------------------------------------------------------------------------
// CDNA5 (gfx1250) implementation. wave32, WMMA f32_16x16x32_f16, LDS-staged.
// ---------------------------------------------------------------------------

typedef __attribute__((ext_vector_type(16))) _Float16 v16h;
typedef __attribute__((ext_vector_type(8)))  _Float16 v8h;
typedef __attribute__((ext_vector_type(8)))  float    v8f;

union F16x16 { v16h v; v8h h8[2]; };

__device__ __forceinline__ float lrelu_f(float v) { return v > 0.f ? v : 0.01f * v; }
__device__ __forceinline__ int imin(int a, int b) { return a < b ? a : b; }
__device__ __forceinline__ int imax(int a, int b) { return a > b ? a : b; }

// ---------------------------------------------------------------------------
// Generic 3x3 conv (VALU) : circular or zero pad, optional BN, residual, lrelu.
// Used for the small-channel encoder / q / k / spe-v convs.
// ---------------------------------------------------------------------------
__global__ void k_conv3x3(const float* __restrict__ in, const float* __restrict__ wgt,
                          const float* __restrict__ bias,
                          const float* __restrict__ bng, const float* __restrict__ bnb,
                          const float* resid, float* out,
                          int N, int Cin, int Cout, int H, int W,
                          int circular, int do_lrelu, int total) {
    int idx = blockIdx.x * blockDim.x + threadIdx.x;
    if (idx >= total) return;
    int x  = idx % W;
    int y  = (idx / W) % H;
    int co = (idx / (W * H)) % Cout;
    int n  = idx / (W * H * Cout);
    float acc = bias[co];
    const float* wp = wgt + (size_t)co * Cin * 9;
    for (int ci = 0; ci < Cin; ++ci) {
        const float* ip = in + ((size_t)(n * Cin + ci) * H) * W;
        for (int ky = 0; ky < 3; ++ky) {
            int yy = y + ky - 1;
            if (circular) yy = (yy + H) % H;
            else if (yy < 0 || yy >= H) continue;
            for (int kx = 0; kx < 3; ++kx) {
                int xx = x + kx - 1;
                if (circular) xx = (xx + W) % W;
                else if (xx < 0 || xx >= W) continue;
                acc += ip[yy * W + xx] * wp[ci * 9 + ky * 3 + kx];
            }
        }
    }
    if (bng) {
        const float inv = 0.99999500003749973f;  // 1/sqrt(1+1e-5)
        acc = acc * (bng[co] * inv) + bnb[co];
    }
    if (resid) acc += resid[idx];
    if (do_lrelu) acc = lrelu_f(acc);
    out[idx] = acc;
}

// 4x4 conv, stride 4, pad 2, 32x32 -> 9x9, fused lrelu  (spe q / k convs)
__global__ void k_conv4x4s4(const float* __restrict__ in, const float* __restrict__ wgt,
                            const float* __restrict__ bias, float* __restrict__ out,
                            int N, int Cin, int Cout, int Hin, int total) {
    int idx = blockIdx.x * blockDim.x + threadIdx.x;
    if (idx >= total) return;
    int ox = idx % 9;
    int oy = (idx / 9) % 9;
    int co = (idx / 81) % Cout;
    int n  = idx / (81 * Cout);
    float acc = bias[co];
    for (int ci = 0; ci < Cin; ++ci) {
        const float* ip = in + ((size_t)(n * Cin + ci) * Hin) * Hin;
        const float* wp = wgt + (((size_t)co * Cin + ci) * 16);
        for (int ky = 0; ky < 4; ++ky) {
            int yy = oy * 4 + ky - 2;
            if (yy < 0 || yy >= Hin) continue;
            for (int kx = 0; kx < 4; ++kx) {
                int xx = ox * 4 + kx - 2;
                if (xx < 0 || xx >= Hin) continue;
                acc += ip[yy * Hin + xx] * wp[ky * 4 + kx];
            }
        }
    }
    out[idx] = lrelu_f(acc);
}

// down16: bilinear 1/16 == mean of 2x2 pixels at offsets (7,8)
__global__ void k_down16(const float* __restrict__ ms, float* __restrict__ out, int total) {
    int idx = blockIdx.x * blockDim.x + threadIdx.x;
    if (idx >= total) return;
    int x = idx % 32, y = (idx / 32) % 32, c = (idx / 1024) % 4, n = idx / 4096;
    const float* p = ms + ((size_t)(n * 4 + c) * 512) * 512;
    int y0 = 16 * y + 7, x0 = 16 * x + 7;
    out[idx] = 0.25f * (p[y0 * 512 + x0] + p[y0 * 512 + x0 + 1] +
                        p[(y0 + 1) * 512 + x0] + p[(y0 + 1) * 512 + x0 + 1]);
}

// upsample16 (bilinear, half-pixel centers, edge clamp) -> initializes d_out
__global__ void k_up16(const float* __restrict__ hs, float* __restrict__ out, int total) {
    int idx = blockIdx.x * blockDim.x + threadIdx.x;
    if (idx >= total) return;
    int x = idx % 512, y = (idx / 512) % 512, c = (idx >> 18) & 127, n = idx >> 25;
    float fy = (y + 0.5f) * 0.0625f - 0.5f;
    float fx = (x + 0.5f) * 0.0625f - 0.5f;
    int y0f = (int)floorf(fy), x0f = (int)floorf(fx);
    float wy = fy - y0f, wx = fx - x0f;
    int y0 = imin(imax(y0f, 0), 31), y1 = imin(imax(y0f + 1, 0), 31);
    int x0 = imin(imax(x0f, 0), 31), x1 = imin(imax(x0f + 1, 0), 31);
    const float* p = hs + ((size_t)(n * 128 + c) * 32) * 32;
    float v = (1.f - wy) * ((1.f - wx) * p[y0 * 32 + x0] + wx * p[y0 * 32 + x1]) +
              wy * ((1.f - wx) * p[y1 * 32 + x0] + wx * p[y1 * 32 + x1]);
    out[idx] = v;
}

// pack conv weights [Cout][Cin][3][3] f32 -> [Cout][tap*Cin + ci] f16 (K-major)
__global__ void k_pack_w(const float* __restrict__ w, _Float16* __restrict__ wT,
                         int Cin, int total) {
    int idx = blockIdx.x * blockDim.x + threadIdx.x;
    if (idx >= total) return;
    int t  = idx % 9;
    int ci = (idx / 9) % Cin;
    int co = idx / (9 * Cin);
    wT[(size_t)co * Cin * 9 + t * Cin + ci] = (_Float16)w[idx];
}

// hs NCHW f32 -> NHWC f16
__global__ void k_cvt_hs(const float* __restrict__ hs, _Float16* __restrict__ act, int total) {
    int idx = blockIdx.x * blockDim.x + threadIdx.x;
    if (idx >= total) return;
    int c = idx & 127, x = (idx >> 7) & 31, y = (idx >> 12) & 31, n = idx >> 17;
    act[idx] = (_Float16)hs[((size_t)(n * 128 + c) * 32 + y) * 32 + x];
}

// ---------------------------------------------------------------------------
// WMMA implicit-GEMM 3x3 conv, Cin=Cout=128, zero pad, NHWC f16 activations.
// 8 waves/block; LDS-staged activation halo [3][34][128]; each wave computes
// TWO 16x16 D tiles (32 x-positions x 128 couts per block), sharing B frags.
// ---------------------------------------------------------------------------
__global__ __launch_bounds__(256)
void k_wmma_conv3x3_c128(const _Float16* __restrict__ act, const _Float16* __restrict__ wT,
                         const float* __restrict__ bias, float* out,
                         int H, int W, float scale, int accumulate, int do_lrelu) {
    __shared__ _Float16 tileA[3 * 34 * 128];   // 26112 B
    const int tid  = threadIdx.x;
    const int wid  = tid >> 5;
    const int lane = tid & 31;
    const int g    = lane >> 4;       // half-group -> K offset select
    const int c    = lane & 15;       // A row / B column within a tile
    const int wtiles = W >> 5;        // 32 x-positions per block
    const int bid = blockIdx.x;
    const int x0 = (bid % wtiles) * 32;
    const int y  = (bid / wtiles) % H;
    const int n  = bid / (wtiles * H);

    // stage activation halo with zero padding: rows y-1..y+1, x0-1..x0+32
    for (int chunk = tid; chunk < 3 * 34 * 8; chunk += 256) {
        int ky  = chunk / (34 * 8);
        int rem = chunk - ky * (34 * 8);
        int xl  = rem >> 3;
        int cb8 = rem & 7;
        int yy = y + ky - 1;
        int xx = x0 + xl - 1;
        uint4 v = make_uint4(0u, 0u, 0u, 0u);
        if (yy >= 0 && yy < H && xx >= 0 && xx < W)
            v = *(const uint4*)(act + (((size_t)n * H + yy) * W + xx) * 128 + cb8 * 16);
        *(uint4*)(tileA + ((ky * 34 + xl) * 128 + cb8 * 16)) = v;
    }
    __syncthreads();

    const int cout = wid * 16 + c;
    const _Float16* wbase = wT + (size_t)cout * 1152;
    v8f acc0 = {};
    v8f acc1 = {};
    for (int t = 0; t < 9; ++t) {
        const int ky = t / 3, kx = t % 3;
        if (t < 8) __builtin_prefetch(wbase + (t + 1) * 128, 0, 1);  // global_prefetch_b8
        const int p0 = (ky * 34 + c + kx) * 128;                     // tile0 A row in LDS
        for (int cb = 0; cb < 4; ++cb) {
            const int k0 = cb * 32 + 8 * g;
            F16x16 a0, a1, b;
            const _Float16* wp = wbase + t * 128 + k0;
            b.h8[0]  = *(const v8h*)(wp);
            b.h8[1]  = *(const v8h*)(wp + 16);
            a0.h8[0] = *(const v8h*)(tileA + p0 + k0);
            a0.h8[1] = *(const v8h*)(tileA + p0 + k0 + 16);
            a1.h8[0] = *(const v8h*)(tileA + p0 + 16 * 128 + k0);
            a1.h8[1] = *(const v8h*)(tileA + p0 + 16 * 128 + k0 + 16);
            acc0 = __builtin_amdgcn_wmma_f32_16x16x32_f16(false, a0.v, false, b.v,
                                                          (short)0, acc0, false, false);
            acc1 = __builtin_amdgcn_wmma_f32_16x16x32_f16(false, a1.v, false, b.v,
                                                          (short)0, acc1, false, false);
        }
    }
    const float bv = bias[cout];
    for (int r = 0; r < 8; ++r) {
        const int m = r + 8 * g;           // C/D layout: M = vgpr + 8*(lane>=16)
        float v0 = acc0[r] + bv;
        float v1 = acc1[r] + bv;
        if (do_lrelu) { v0 = lrelu_f(v0); v1 = lrelu_f(v1); }
        v0 *= scale; v1 *= scale;
        size_t o0 = (((size_t)n * 128 + cout) * H + y) * W + (x0 + m);
        size_t o1 = o0 + 16;
        if (accumulate) { out[o0] += v0; out[o1] += v1; }
        else            { out[o0] = v0;  out[o1] = v1; }
    }
}

// ---------------------------------------------------------------------------
// Attention patchify / normalization helpers
// ---------------------------------------------------------------------------
// qA f16 [n][65536][64]  (K=40 padded with zeros)
__global__ void k_patch_q(const float* __restrict__ qf, _Float16* __restrict__ qA, int total) {
    int idx = blockIdx.x * blockDim.x + threadIdx.x;
    if (idx >= total) return;
    int cc = idx & 63, q = (idx >> 6) & 65535, n = idx >> 22;
    _Float16 v = (_Float16)0.f;
    if (cc < 40) {
        int ch = cc >> 2, a = (cc >> 1) & 1, b = cc & 1;
        int i = q >> 8, j = q & 255;
        v = (_Float16)qf[(((size_t)(n * 10 + ch)) * 512 + (2 * i + a)) * 512 + (2 * j + b)];
    }
    qA[idx] = v;
}

// kpack f32 [n][256][40] with roll
__global__ void k_patch_k(const float* __restrict__ kf, float* __restrict__ kpack,
                          int sy, int sx, int total) {
    int idx = blockIdx.x * blockDim.x + threadIdx.x;
    if (idx >= total) return;
    int cc = idx % 40, l = (idx / 40) & 255, n = idx / (40 * 256);
    int ch = cc >> 2, a = (cc >> 1) & 1, b = cc & 1;
    int i = l >> 4, j = l & 15;
    int y = (2 * i + a - sy) & 31, x = (2 * j + b - sx) & 31;
    kpack[idx] = kf[((size_t)(n * 10 + ch) * 32 + y) * 32 + x];
}

// per-image kmax = max_l ||kp[:,l]||
__global__ void k_knorm(const float* __restrict__ kpack, float* __restrict__ kmaxn) {
    __shared__ float sm[256];
    int n = blockIdx.x, l = threadIdx.x;
    const float* p = kpack + ((size_t)n * 256 + l) * 40;
    float s = 0.f;
    for (int cc = 0; cc < 40; ++cc) s += p[cc] * p[cc];
    sm[l] = sqrtf(s);
    __syncthreads();
    for (int off = 128; off; off >>= 1) {
        if (l < off) sm[l] = fmaxf(sm[l], sm[l + off]);
        __syncthreads();
    }
    if (l == 0) kmaxn[n] = sm[0];
}

// kpB f16 [n][256][64]
__global__ void k_kpb(const float* __restrict__ kpack, const float* __restrict__ kmaxn,
                      _Float16* __restrict__ kpB, int total) {
    int idx = blockIdx.x * blockDim.x + threadIdx.x;
    if (idx >= total) return;
    int cc = idx & 63, l = (idx >> 6) & 255, n = idx >> 14;
    _Float16 v = (_Float16)0.f;
    if (cc < 40)
        v = (_Float16)(kpack[((size_t)n * 256 + l) * 40 + cc] / (kmaxn[n] + 1e-5f));
    kpB[idx] = v;
}

// vp f16 [n][512][256] with roll (B operand of GEMM2: column d, K=l contiguous)
__global__ void k_patch_v(const float* __restrict__ vf, _Float16* __restrict__ vp,
                          int sy, int sx, int total) {
    int idx = blockIdx.x * blockDim.x + threadIdx.x;
    if (idx >= total) return;
    int l = idx & 255, d = (idx >> 8) & 511, n = idx >> 17;
    int C = d >> 2, a = (d >> 1) & 1, b = d & 1;
    int i = l >> 4, j = l & 15;
    int y = (2 * i + a - sy) & 31, x = (2 * j + b - sx) & 31;
    vp[idx] = (_Float16)vf[((size_t)(n * 128 + C) * 32 + y) * 32 + x];
}

// ---------------------------------------------------------------------------
// WMMA GEMM1: S[q][l] = 10 * sum_c qA[q][c] * kpB[l][c]   (M=65536,N=256,K=64)
// All 8 waves of a block share the same M-tile -> LDS-staged A.
// ---------------------------------------------------------------------------
__global__ __launch_bounds__(256)
void k_gemm_qk(const _Float16* __restrict__ qA, const _Float16* __restrict__ kpB,
               float* __restrict__ S) {
    __shared__ _Float16 tA[16 * 64];   // 2 KB
    const int n = blockIdx.y;
    const int tid = threadIdx.x, wid = tid >> 5, lane = tid & 31;
    const int g = lane >> 4, c = lane & 15;
    const int mt = blockIdx.x >> 1;                    // (blockIdx.x*8 + wid) >> 4, wave-invariant
    const int nt = (blockIdx.x * 8 + wid) & 15;
    if (tid < 64) {
        int row = tid >> 2, seg = tid & 3;
        *(uint4*)(tA + row * 64 + seg * 16) =
            *(const uint4*)(qA + ((size_t)n * 65536 + mt * 16 + row) * 64 + seg * 16);
    }
    __syncthreads();
    const _Float16* brow = kpB + ((size_t)n * 256 + nt * 16 + c) * 64;
    v8f acc = {};
    for (int kk = 0; kk < 2; ++kk) {
        const int k0 = kk * 32 + 8 * g;
        F16x16 a, b;
        a.h8[0] = *(const v8h*)(tA + c * 64 + k0);
        a.h8[1] = *(const v8h*)(tA + c * 64 + k0 + 16);
        b.h8[0] = *(const v8h*)(brow + k0);
        b.h8[1] = *(const v8h*)(brow + k0 + 16);
        acc = __builtin_amdgcn_wmma_f32_16x16x32_f16(false, a.v, false, b.v,
                                                     (short)0, acc, false, false);
    }
    float* srow = S + ((size_t)n * 65536 + mt * 16) * 256 + nt * 16 + c;
    for (int r = 0; r < 8; ++r) {
        const int m = r + 8 * g;
        srow[(size_t)m * 256] = acc[r] * 10.f;
    }
}

// wave-per-row softmax over 256 cols, f32 -> f16
__global__ void k_softmax_rows(const float* __restrict__ S, _Float16* __restrict__ P, int rows) {
    int wid = (blockIdx.x * blockDim.x + threadIdx.x) >> 5;
    int lane = threadIdx.x & 31;
    if (wid >= rows) return;
    const float* s = S + (size_t)wid * 256;
    float vals[8], m = -1e30f;
    for (int t = 0; t < 8; ++t) { vals[t] = s[lane + 32 * t]; m = fmaxf(m, vals[t]); }
    for (int off = 16; off; off >>= 1) m = fmaxf(m, __shfl_xor(m, off, 32));
    float sum = 0.f;
    for (int t = 0; t < 8; ++t) { vals[t] = __expf(vals[t] - m); sum += vals[t]; }
    for (int off = 16; off; off >>= 1) sum += __shfl_xor(sum, off, 32);
    float r = 1.f / sum;
    _Float16* p = P + (size_t)wid * 256;
    for (int t = 0; t < 8; ++t) p[lane + 32 * t] = (_Float16)(vals[t] * r);
}

// ---------------------------------------------------------------------------
// WMMA GEMM2: out[q][d] = (1/6) sum_l P[q][l] vp[d][l], stored folded as NHWC
// f16 (M=65536, N=512, K=256). 8 waves share one M-tile -> LDS-staged A.
// ---------------------------------------------------------------------------
__global__ __launch_bounds__(256)
void k_gemm_av(const _Float16* __restrict__ P, const _Float16* __restrict__ vp,
               _Float16* __restrict__ act) {
    __shared__ _Float16 tA[16 * 256];  // 8 KB
    const int n = blockIdx.y;
    const int tid = threadIdx.x, wid = tid >> 5, lane = tid & 31;
    const int g = lane >> 4, c = lane & 15;
    const int mt = blockIdx.x >> 2;                    // (blockIdx.x*8 + wid) >> 5, wave-invariant
    const int nt = (blockIdx.x * 8 + wid) & 31;
    {
        int row = tid >> 4, seg = tid & 15;
        *(uint4*)(tA + row * 256 + seg * 16) =
            *(const uint4*)(P + ((size_t)n * 65536 + mt * 16 + row) * 256 + seg * 16);
    }
    __syncthreads();
    const _Float16* brow = vp + ((size_t)n * 512 + nt * 16 + c) * 256;
    v8f acc = {};
    for (int kk = 0; kk < 8; ++kk) {
        const int k0 = kk * 32 + 8 * g;
        F16x16 a, b;
        a.h8[0] = *(const v8h*)(tA + c * 256 + k0);
        a.h8[1] = *(const v8h*)(tA + c * 256 + k0 + 16);
        b.h8[0] = *(const v8h*)(brow + k0);
        b.h8[1] = *(const v8h*)(brow + k0 + 16);
        acc = __builtin_amdgcn_wmma_f32_16x16x32_f16(false, a.v, false, b.v,
                                                     (short)0, acc, false, false);
    }
    for (int r = 0; r < 8; ++r) {
        const int m = r + 8 * g;
        const int q = mt * 16 + m;
        const int i = q >> 8, j = q & 255;
        const int d = nt * 16 + c;
        const int C = d >> 2, a = (d >> 1) & 1, b = d & 1;
        const int y = 2 * i + a, x = 2 * j + b;
        act[(((size_t)n * 512 + y) * 512 + x) * 128 + C] = (_Float16)(acc[r] * (1.f / 6.f));
    }
}

// ---------------------------------------------------------------------------
// spe attention small kernels
// ---------------------------------------------------------------------------
__global__ void k_spe_qmax(const float* __restrict__ q9, float* __restrict__ kscal) {
    __shared__ float sm[8];
    int t = threadIdx.x;
    if (t < 8) {
        const float* p = q9 + t * 81;
        float s = 0.f;
        for (int i = 0; i < 81; ++i) s += p[i] * p[i];
        sm[t] = sqrtf(s);
    }
    __syncthreads();
    if (t == 0) {
        float m = sm[0];
        for (int i = 1; i < 8; ++i) m = fmaxf(m, sm[i]);
        kscal[0] = m;
    }
}

__global__ void k_spe_score(const float* __restrict__ q9, const float* __restrict__ k9,
                            const float* __restrict__ kscal, float* __restrict__ S4, int total) {
    int idx = blockIdx.x * blockDim.x + threadIdx.x;
    if (idx >= total) return;
    int m = idx % 128, j = (idx / 128) % 4, n = idx / 512;
    float inv = 1.f / kscal[0];
    const float* qp = q9 + (n * 4 + j) * 81;
    const float* kp = k9 + (n * 128 + m) * 81;
    float s = 0.f;
    for (int p = 0; p < 81; ++p) s += qp[p] * kp[p];
    S4[(n * 4 + j) * 128 + m] = s * inv * 10.f;
}

__global__ void k_spe_softmax(const float* __restrict__ S4, float* __restrict__ attw, int total) {
    int idx = blockIdx.x * blockDim.x + threadIdx.x;
    if (idx >= total) return;
    int m = idx % 128, n = idx / 128;
    float v[4], mx = -1e30f;
    for (int j = 0; j < 4; ++j) { v[j] = S4[(n * 4 + j) * 128 + m]; mx = fmaxf(mx, v[j]); }
    float sum = 0.f;
    for (int j = 0; j < 4; ++j) { v[j] = __expf(v[j] - mx); sum += v[j]; }
    for (int j = 0; j < 4; ++j) attw[(n * 4 + j) * 128 + m] = v[j] / sum;
}

// act[nhwc] = (1/6) sum_j attw[n][j][c] * hrms[n][j][y][x]
__global__ void k_spe_mix(const float* __restrict__ attw, const float* __restrict__ hrms,
                          _Float16* __restrict__ act, int total) {
    int idx = blockIdx.x * blockDim.x + threadIdx.x;
    if (idx >= total) return;
    int c = idx & 127, x = (idx >> 7) & 511, y = (idx >> 16) & 511, n = idx >> 25;
    float s = 0.f;
    for (int j = 0; j < 4; ++j)
        s += attw[(n * 4 + j) * 128 + c] * hrms[((size_t)(n * 4 + j) * 512 + y) * 512 + x];
    act[idx] = (_Float16)(s * (1.f / 6.f));
}

// ---------------------------------------------------------------------------
// Host side
// ---------------------------------------------------------------------------
extern "C" void kernel_launch(void* const* d_in, const int* in_sizes, int n_in,
                              void* d_out, int out_size, void* d_ws, size_t ws_size,
                              hipStream_t stream) {
    (void)in_sizes; (void)n_in; (void)out_size; (void)ws_size;
    const int TPB = 256;
    const float* hs = (const float*)d_in[0];
    const float* ms = (const float*)d_in[1];

    // params (alphabetical pytree order): att[4]{k.b,k.w,q.b,q.w,r.b,r.w,v.b,v.w},
    // enc_hs{b1,b2,c1.b,c1.w,c2.b,c2.w,g1,g2,res[5]{c1.b,c1.w,c2.b,c2.w}}, enc_ms same, spe{k,q,r,v}.
    const float *h_kb[4], *h_kw[4], *h_qb[4], *h_qw[4], *h_rb[4], *h_rw[4], *h_vb[4], *h_vw[4];
    for (int i = 0; i < 4; ++i) {
        int b = 2 + i * 8;
        h_kb[i] = (const float*)d_in[b + 0]; h_kw[i] = (const float*)d_in[b + 1];
        h_qb[i] = (const float*)d_in[b + 2]; h_qw[i] = (const float*)d_in[b + 3];
        h_rb[i] = (const float*)d_in[b + 4]; h_rw[i] = (const float*)d_in[b + 5];
        h_vb[i] = (const float*)d_in[b + 6]; h_vw[i] = (const float*)d_in[b + 7];
    }
    struct EncP {
        const float *b1, *b2, *c1b, *c1w, *c2b, *c2w, *g1, *g2;
        const float *rc1b[5], *rc1w[5], *rc2b[5], *rc2w[5];
    };
    auto load_enc = [&](int base) {
        EncP e;
        e.b1 = (const float*)d_in[base + 0]; e.b2 = (const float*)d_in[base + 1];
        e.c1b = (const float*)d_in[base + 2]; e.c1w = (const float*)d_in[base + 3];
        e.c2b = (const float*)d_in[base + 4]; e.c2w = (const float*)d_in[base + 5];
        e.g1 = (const float*)d_in[base + 6]; e.g2 = (const float*)d_in[base + 7];
        for (int r = 0; r < 5; ++r) {
            e.rc1b[r] = (const float*)d_in[base + 8 + r * 4 + 0];
            e.rc1w[r] = (const float*)d_in[base + 8 + r * 4 + 1];
            e.rc2b[r] = (const float*)d_in[base + 8 + r * 4 + 2];
            e.rc2w[r] = (const float*)d_in[base + 8 + r * 4 + 3];
        }
        return e;
    };
    EncP ehs = load_enc(34);
    EncP ems = load_enc(62);
    const float* spe_kb = (const float*)d_in[90]; const float* spe_kw = (const float*)d_in[91];
    const float* spe_qb = (const float*)d_in[92]; const float* spe_qw = (const float*)d_in[93];
    const float* spe_rb = (const float*)d_in[94]; const float* spe_rw = (const float*)d_in[95];
    const float* spe_vb = (const float*)d_in[96]; const float* spe_vw = (const float*)d_in[97];

    // ---- workspace carve ----
    char* wp = (char*)d_ws;
    auto carve = [&](size_t bytes) -> void* {
        void* p = (void*)wp;
        wp += (bytes + 255) & ~(size_t)255;
        return p;
    };
    _Float16* hsNHWC = (_Float16*)carve((size_t)2 * 32 * 32 * 128 * 2);
    _Float16* wTv[4]; _Float16* wTr[4];
    for (int i = 0; i < 4; ++i) wTv[i] = (_Float16*)carve((size_t)128 * 1152 * 2);
    for (int i = 0; i < 4; ++i) wTr[i] = (_Float16*)carve((size_t)128 * 1152 * 2);
    _Float16* wTspe = (_Float16*)carve((size_t)128 * 1152 * 2);
    float* eh_a = (float*)carve((size_t)2 * 10 * 1024 * 4);
    float* eh_b = (float*)carve((size_t)2 * 10 * 1024 * 4);
    float* em_a = (float*)carve((size_t)2 * 10 * 262144 * 4);   // also reused as qf
    float* em_b = (float*)carve((size_t)2 * 10 * 262144 * 4);   // fea_ms
    float* ms16 = (float*)carve((size_t)2 * 4 * 1024 * 4);
    float* q9   = (float*)carve((size_t)2 * 4 * 81 * 4);
    float* k9   = (float*)carve((size_t)2 * 128 * 81 * 4);
    float* kscal = (float*)carve(256);
    float* S4   = (float*)carve((size_t)2 * 4 * 128 * 4);
    float* attw = (float*)carve((size_t)2 * 4 * 128 * 4);
    float* hrms = (float*)carve((size_t)2 * 4 * 262144 * 4);
    _Float16* actb = (_Float16*)carve((size_t)2 * 512 * 512 * 128 * 2);
    float* kpack = (float*)carve((size_t)2 * 256 * 40 * 4);
    float* kmaxn = (float*)carve(256);
    _Float16* kpB = (_Float16*)carve((size_t)2 * 256 * 64 * 2);
    _Float16* qA  = (_Float16*)carve((size_t)2 * 65536 * 64 * 2);
    float* vf = (float*)carve((size_t)2 * 128 * 1024 * 4);
    _Float16* vp = (_Float16*)carve((size_t)2 * 512 * 256 * 2);
    float* Sbuf = (float*)carve((size_t)2 * 65536 * 256 * 4);
    _Float16* Pbuf = (_Float16*)carve((size_t)2 * 65536 * 256 * 2);
    float* outF = (float*)d_out;

#define GRID(tot) dim3(((tot) + TPB - 1) / TPB)

    // ---- precompute: hs NHWC f16, packed f16 weights ----
    k_cvt_hs<<<GRID(262144), TPB, 0, stream>>>(hs, hsNHWC, 262144);
    for (int i = 0; i < 4; ++i) {
        k_pack_w<<<GRID(147456), TPB, 0, stream>>>(h_vw[i], wTv[i], 128, 147456);
        k_pack_w<<<GRID(147456), TPB, 0, stream>>>(h_rw[i], wTr[i], 128, 147456);
    }
    k_pack_w<<<GRID(147456), TPB, 0, stream>>>(spe_rw, wTspe, 128, 147456);

    // ---- encoder(hs): circular convs on 32x32 ----
    k_conv3x3<<<GRID(8192), TPB, 0, stream>>>(hs, ehs.c1w, ehs.c1b, ehs.g1, ehs.b1, nullptr,
                                              eh_a, 2, 128, 4, 32, 32, 1, 1, 8192);
    k_conv3x3<<<GRID(20480), TPB, 0, stream>>>(eh_a, ehs.c2w, ehs.c2b, ehs.g2, ehs.b2, nullptr,
                                               eh_b, 2, 4, 10, 32, 32, 1, 0, 20480);
    for (int r = 0; r < 5; ++r) {
        k_conv3x3<<<GRID(20480), TPB, 0, stream>>>(eh_b, ehs.rc1w[r], ehs.rc1b[r], nullptr, nullptr,
                                                   nullptr, eh_a, 2, 10, 10, 32, 32, 1, 1, 20480);
        k_conv3x3<<<GRID(20480), TPB, 0, stream>>>(eh_a, ehs.rc2w[r], ehs.rc2b[r], nullptr, nullptr,
                                                   eh_b, eh_b, 2, 10, 10, 32, 32, 1, 0, 20480);
    }
    float* fea_hs = eh_b;

    // ---- encoder(ms): circular convs on 512x512 ----
    k_conv3x3<<<GRID(1048576), TPB, 0, stream>>>(ms, ems.c1w, ems.c1b, ems.g1, ems.b1, nullptr,
                                                 em_a, 2, 4, 2, 512, 512, 1, 1, 1048576);
    k_conv3x3<<<GRID(5242880), TPB, 0, stream>>>(em_a, ems.c2w, ems.c2b, ems.g2, ems.b2, nullptr,
                                                 em_b, 2, 2, 10, 512, 512, 1, 0, 5242880);
    for (int r = 0; r < 5; ++r) {
        k_conv3x3<<<GRID(5242880), TPB, 0, stream>>>(em_b, ems.rc1w[r], ems.rc1b[r], nullptr,
                                                     nullptr, nullptr, em_a, 2, 10, 10, 512, 512,
                                                     1, 1, 5242880);
        k_conv3x3<<<GRID(5242880), TPB, 0, stream>>>(em_a, ems.rc2w[r], ems.rc2b[r], nullptr,
                                                     nullptr, em_b, em_b, 2, 10, 10, 512, 512,
                                                     1, 0, 5242880);
    }
    float* fea_ms = em_b;

    // ---- init d_out with upsample16(hs) ----
    k_up16<<<GRID(67108864), TPB, 0, stream>>>(hs, outF, 67108864);

    // ---- spe attention ----
    k_down16<<<GRID(8192), TPB, 0, stream>>>(ms, ms16, 8192);
    k_conv4x4s4<<<GRID(648), TPB, 0, stream>>>(ms16, spe_qw, spe_qb, q9, 2, 4, 4, 32, 648);
    k_conv4x4s4<<<GRID(20736), TPB, 0, stream>>>(hs, spe_kw, spe_kb, k9, 2, 128, 128, 32, 20736);
    k_spe_qmax<<<1, 64, 0, stream>>>(q9, kscal);
    k_spe_score<<<GRID(1024), TPB, 0, stream>>>(q9, k9, kscal, S4, 1024);
    k_spe_softmax<<<GRID(256), TPB, 0, stream>>>(S4, attw, 256);
    k_conv3x3<<<GRID(2097152), TPB, 0, stream>>>(ms, spe_vw, spe_vb, nullptr, nullptr, nullptr,
                                                 hrms, 2, 4, 4, 512, 512, 0, 1, 2097152);
    k_spe_mix<<<GRID(67108864), TPB, 0, stream>>>(attw, hrms, actb, 67108864);
    k_wmma_conv3x3_c128<<<2 * 512 * 16, 256, 0, stream>>>(actb, wTspe, spe_rb, outF,
                                                          512, 512, 1.0f, 1, 1);

    // ---- 4 cross-scale attention heads ----
    const int SY[4] = {0, 0, 1, 1};
    const int SX[4] = {0, 1, 0, 1};
    for (int hI = 0; hI < 4; ++hI) {
        // k path: conv k on fea_hs, roll, patchify, normalize
        k_conv3x3<<<GRID(20480), TPB, 0, stream>>>(fea_hs, h_kw[hI], h_kb[hI], nullptr, nullptr,
                                                   nullptr, eh_a, 2, 10, 10, 32, 32, 0, 1, 20480);
        k_patch_k<<<GRID(20480), TPB, 0, stream>>>(eh_a, kpack, SY[hI], SX[hI], 20480);
        k_knorm<<<2, 256, 0, stream>>>(kpack, kmaxn);
        k_kpb<<<GRID(32768), TPB, 0, stream>>>(kpack, kmaxn, kpB, 32768);
        // q path: conv q on fea_ms, patchify
        k_conv3x3<<<GRID(5242880), TPB, 0, stream>>>(fea_ms, h_qw[hI], h_qb[hI], nullptr, nullptr,
                                                     nullptr, em_a, 2, 10, 10, 512, 512, 0, 1,
                                                     5242880);
        k_patch_q<<<GRID(8388608), TPB, 0, stream>>>(em_a, qA, 8388608);
        // v path: WMMA conv 128->128 on 32x32, roll + patchify
        k_wmma_conv3x3_c128<<<2 * 32 * 1, 256, 0, stream>>>(hsNHWC, wTv[hI], h_vb[hI], vf,
                                                            32, 32, 1.0f, 0, 1);
        k_patch_v<<<GRID(262144), TPB, 0, stream>>>(vf, vp, SY[hI], SX[hI], 262144);
        // attention: scores -> softmax -> weighted sum folded into NHWC f16
        k_gemm_qk<<<dim3(8192, 2), 256, 0, stream>>>(qA, kpB, Sbuf);
        k_softmax_rows<<<GRID(131072 * 32), TPB, 0, stream>>>(Sbuf, Pbuf, 131072);
        k_gemm_av<<<dim3(16384, 2), 256, 0, stream>>>(Pbuf, vp, actb);
        // final r conv, accumulate 0.25 * lrelu(conv) into d_out
        k_wmma_conv3x3_c128<<<2 * 512 * 16, 256, 0, stream>>>(actb, wTr[hI], h_rb[hI], outF,
                                                              512, 512, 0.25f, 1, 1);
    }
#undef GRID
}
